// SeqClassifier_35192962023461
// MI455X (gfx1250) — matrix operand order repaired and checked by reading, not compiled, not run on hardware
//
#include <hip/hip_runtime.h>

#define DEV __device__ __forceinline__

typedef _Float16 vh8  __attribute__((ext_vector_type(8)));
typedef _Float16 vh16 __attribute__((ext_vector_type(16)));
typedef float    vf8  __attribute__((ext_vector_type(8)));

constexpr int SEQ   = 2048;
constexpr int DIM   = 768;
constexpr int NH    = 12;
constexpr int DHD   = 64;
constexpr int FF    = 3072;
constexpr int NG    = 16;    // global tokens
constexpr int CHUNK = 256;
constexpr int NCH   = 8;
constexpr int WIN   = 256;   // W2
constexpr int KWW   = 768;   // CHUNK + 2*WIN
constexpr int TW    = 800;   // NG + KWW padded to multiple of 32

// ---------------- WMMA helpers (CDNA5, wave32) ----------------

DEV vf8 wmma32(vh16 a, vh16 b, vf8 c) {
  // v_wmma_f32_16x16x32_f16 : D = A(16x32 f16) x B(32x16 f16) + C(16x16 f32)
  return __builtin_amdgcn_wmma_f32_16x16x32_f16(false, a, false, b, (short)0, c,
                                                false, false);
}

DEV vh16 zero_vh16() {
  vh16 r;
#pragma unroll
  for (int i = 0; i < 16; ++i) r[i] = (_Float16)0.0f;
  return r;
}

// B fragment: 16 K-contiguous halves starting at p (col = lane&15, kb = (lane>>4)*16)
DEV vh16 load16(const _Float16* p) {
  vh8 lo = *(const vh8*)p;
  vh8 hi = *(const vh8*)(p + 8);
  vh16 r;
#pragma unroll
  for (int i = 0; i < 8; ++i) { r[i] = lo[i]; r[8 + i] = hi[i]; }
  return r;
}

// A fragment: 16 K-contiguous halves at p (K = koff..koff+15)
DEV vh16 loadA16(const _Float16* p) {
  vh8 lo = *(const vh8*)p;
  vh8 hi = *(const vh8*)(p + 8);
  vh16 r;
#pragma unroll
  for (int i = 0; i < 8; ++i) { r[i] = lo[i]; r[8 + i] = hi[i]; }
  return r;
}

DEV float gelu_tanh(float x) {
  const float c0 = 0.7978845608028654f;  // sqrt(2/pi)
  float t = tanhf(c0 * (x + 0.044715f * x * x * x));
  return 0.5f * x * (1.0f + t);
}

// ---------------- weight convert + transpose (fp32 [K,N] -> f16 [N,K]) ----------------

__global__ __launch_bounds__(256) void convT_kernel(const float* __restrict__ W,
                                                    _Float16* __restrict__ Wt,
                                                    int K, int N) {
  int i = blockIdx.x * 256 + threadIdx.x;
  if (i >= K * N) return;
  int k = i / N;
  int n = i - k * N;
  Wt[(size_t)n * K + k] = (_Float16)W[i];
}

// ---------------- generic WMMA GEMM ----------------
// C = act(((A[M,K]f16 x Bt^T) + bias) * alpha) ; Bt stored [N,K] (K contiguous).
// Block = 8 waves; each wave computes a 16x64 strip (4 accumulators).
// K loop unrolled by 64 with all 10 fragment loads issued before the 8 WMMAs,
// so the second k-step's loads overlap the first k-step's matrix ops.
// Requires K % 64 == 0 (true for all call sites: 768, 3072).

__global__ __launch_bounds__(256) void wmma_gemm_kernel(
    const _Float16* __restrict__ A, const _Float16* __restrict__ Bt,
    const float* __restrict__ bias, float* __restrict__ Cout,
    _Float16* __restrict__ Ch, int M, int N, int K, float alpha, int act) {
  const int lane = threadIdx.x & 31;
  const int wave = threadIdx.x >> 5;
  const int n0 = blockIdx.x * 64;
  const int m0 = blockIdx.y * 128 + wave * 16;
  if (m0 >= M) return;

  const int am    = m0 + (lane & 15);
  const int koffA = (lane >> 4) << 4;
  const int koffB = (lane >> 4) << 4;
  const _Float16* aBase = A + (size_t)am * K + koffA;
  const _Float16* bBase[4];
  int bn[4];
#pragma unroll
  for (int j = 0; j < 4; ++j) {
    bn[j] = n0 + j * 16 + (lane & 15);
    bBase[j] = Bt + (size_t)bn[j] * K + koffB;
  }

  vf8 acc[4];
#pragma unroll
  for (int j = 0; j < 4; ++j) acc[j] = (vf8){0.f, 0.f, 0.f, 0.f, 0.f, 0.f, 0.f, 0.f};

  for (int k0 = 0; k0 < K; k0 += 64) {
    __builtin_prefetch(aBase + k0 + 256, 0, 0);
    // issue ALL loads for both 32-deep k-steps before any WMMA
    vh16 a0 = loadA16(aBase + k0);
    vh16 b0 = load16(bBase[0] + k0);
    vh16 b1 = load16(bBase[1] + k0);
    vh16 b2 = load16(bBase[2] + k0);
    vh16 b3 = load16(bBase[3] + k0);
    vh16 a1 = loadA16(aBase + k0 + 32);
    vh16 b4 = load16(bBase[0] + k0 + 32);
    vh16 b5 = load16(bBase[1] + k0 + 32);
    vh16 b6 = load16(bBase[2] + k0 + 32);
    vh16 b7 = load16(bBase[3] + k0 + 32);
    // k-step 0 (k-step 1 loads still in flight)
    acc[0] = wmma32(a0, b0, acc[0]);
    acc[1] = wmma32(a0, b1, acc[1]);
    acc[2] = wmma32(a0, b2, acc[2]);
    acc[3] = wmma32(a0, b3, acc[3]);
    // k-step 1
    acc[0] = wmma32(a1, b4, acc[0]);
    acc[1] = wmma32(a1, b5, acc[1]);
    acc[2] = wmma32(a1, b6, acc[2]);
    acc[3] = wmma32(a1, b7, acc[3]);
  }

  const int rbase = m0 + ((lane >> 4) << 3);
#pragma unroll
  for (int j = 0; j < 4; ++j) {
    const float bv = bias ? bias[bn[j]] : 0.0f;
#pragma unroll
    for (int r = 0; r < 8; ++r) {
      float v = (acc[j][r] + bv) * alpha;
      if (act == 1) v = gelu_tanh(v);
      size_t o = (size_t)(rbase + r) * N + bn[j];
      if (Cout) Cout[o] = v;
      if (Ch)   Ch[o]   = (_Float16)v;
    }
  }
}

// ---------------- embedding gather + LayerNorm ----------------

__global__ __launch_bounds__(256) void embed_ln_kernel(
    const int* __restrict__ ids, const float* __restrict__ etok,
    const float* __restrict__ epos, const float* __restrict__ g,
    const float* __restrict__ b, float* __restrict__ X, _Float16* __restrict__ Xh) {
  __shared__ float red[256];
  __shared__ float s_mean, s_rstd;
  const int s = blockIdx.x, tid = threadIdx.x;
  const int id = ids[s];
  float v[3];
#pragma unroll
  for (int i = 0; i < 3; ++i) {
    int d = tid + i * 256;
    v[i] = etok[(size_t)id * DIM + d] + epos[(size_t)s * DIM + d];
  }
  float lsum = v[0] + v[1] + v[2];
  red[tid] = lsum;
  __syncthreads();
  for (int off = 128; off > 0; off >>= 1) {
    if (tid < off) red[tid] += red[tid + off];
    __syncthreads();
  }
  if (tid == 0) s_mean = red[0] * (1.0f / DIM);
  __syncthreads();
  const float m = s_mean;
  float lv = 0.f;
#pragma unroll
  for (int i = 0; i < 3; ++i) { float d = v[i] - m; lv += d * d; }
  red[tid] = lv;
  __syncthreads();
  for (int off = 128; off > 0; off >>= 1) {
    if (tid < off) red[tid] += red[tid + off];
    __syncthreads();
  }
  if (tid == 0) s_rstd = rsqrtf(red[0] * (1.0f / DIM) + 1e-5f);
  __syncthreads();
  const float r = s_rstd;
#pragma unroll
  for (int i = 0; i < 3; ++i) {
    int d = tid + i * 256;
    float y = (v[i] - m) * r * g[d] + b[d];
    X[(size_t)s * DIM + d] = y;
    Xh[(size_t)s * DIM + d] = (_Float16)y;
  }
}

// ---------------- residual add + LayerNorm (in-place on X) ----------------

__global__ __launch_bounds__(256) void add_ln_kernel(
    float* __restrict__ X, const float* __restrict__ Aadd,
    const float* __restrict__ g, const float* __restrict__ b,
    _Float16* __restrict__ Xh) {
  __shared__ float red[256];
  __shared__ float s_mean, s_rstd;
  const int s = blockIdx.x, tid = threadIdx.x;
  float v[3];
#pragma unroll
  for (int i = 0; i < 3; ++i) {
    int d = tid + i * 256;
    v[i] = X[(size_t)s * DIM + d] + Aadd[(size_t)s * DIM + d];
  }
  float lsum = v[0] + v[1] + v[2];
  red[tid] = lsum;
  __syncthreads();
  for (int off = 128; off > 0; off >>= 1) {
    if (tid < off) red[tid] += red[tid + off];
    __syncthreads();
  }
  if (tid == 0) s_mean = red[0] * (1.0f / DIM);
  __syncthreads();
  const float m = s_mean;
  float lv = 0.f;
#pragma unroll
  for (int i = 0; i < 3; ++i) { float d = v[i] - m; lv += d * d; }
  red[tid] = lv;
  __syncthreads();
  for (int off = 128; off > 0; off >>= 1) {
    if (tid < off) red[tid] += red[tid + off];
    __syncthreads();
  }
  if (tid == 0) s_rstd = rsqrtf(red[0] * (1.0f / DIM) + 1e-5f);
  __syncthreads();
  const float r = s_rstd;
#pragma unroll
  for (int i = 0; i < 3; ++i) {
    int d = tid + i * 256;
    float y = (v[i] - m) * r * g[d] + b[d];
    X[(size_t)s * DIM + d] = y;
    Xh[(size_t)s * DIM + d] = (_Float16)y;
  }
}

// ---------------- banded local + global-column attention (one 16-row q tile) ----------------
// grid = (C/16=16, NC=8, H=12). Scores -> LDS, softmax, P(f16) x V via WMMA.

__global__ __launch_bounds__(256) void attn_chunk_kernel(
    const _Float16* __restrict__ qh, const _Float16* __restrict__ kh,
    const _Float16* __restrict__ vh, _Float16* __restrict__ aoh) {
  __shared__ __align__(16) float sS[16 * TW];   // 51.2 KB
  __shared__ float red[256];
  __shared__ float rowmax[16];
  __shared__ float rowsum[16];

  const int tid  = threadIdx.x;
  const int lane = tid & 31;
  const int wave = tid >> 5;
  const int qt = blockIdx.x;   // 16-row tile within chunk
  const int n  = blockIdx.y;   // chunk index
  const int h  = blockIdx.z;   // head
  const int qrow0 = n * CHUNK + qt * 16;

  // ---- phase 1: scores s[16, TW] (cols: 0..15 global, 16..783 local, 784.. pad) ----
  const int am    = qrow0 + (lane & 15);
  const int koffA = (lane >> 4) << 4;
  const int koffB = (lane >> 4) << 4;
  const _Float16* aBase = qh + (size_t)am * DIM + h * DHD + koffA;
  const vh16 aq0 = loadA16(aBase);        // Q fragment, K = 0..31  (loop-invariant)
  const vh16 aq1 = loadA16(aBase + 32);   // Q fragment, K = 32..63

  for (int t = wave; t < TW / 16; t += 8) {
    const int ct0 = t * 16;
    const int cn  = ct0 + (lane & 15);
    int tok = -1;
    if (cn < NG) tok = cn;
    else if (cn < NG + KWW) {
      int kpos = n * CHUNK + (cn - NG) - WIN;
      if (kpos >= 0 && kpos < SEQ) tok = kpos;
    }
    vf8 acc = {0.f, 0.f, 0.f, 0.f, 0.f, 0.f, 0.f, 0.f};
    const _Float16* bB =
        (tok >= 0) ? (kh + (size_t)tok * DIM + h * DHD + koffB) : nullptr;
    vh16 b0 = bB ? load16(bB) : zero_vh16();
    vh16 b1 = bB ? load16(bB + 32) : zero_vh16();
    acc = wmma32(aq0, b0, acc);
    acc = wmma32(aq1, b1, acc);

    const int rl0 = (lane >> 4) << 3;
#pragma unroll
    for (int r = 0; r < 8; ++r) {
      const int rloc = rl0 + r;
      const int qcl  = qt * 16 + rloc;  // q index within chunk
      bool valid;
      if (cn < NG) valid = true;
      else if (cn < NG + KWW) {
        int cl   = cn - NG;
        int kpos = n * CHUNK + cl - WIN;
        int dd   = cl - WIN - qcl;
        valid = (dd <= WIN) && (dd >= -WIN) && (kpos >= NG) && (kpos < SEQ);
      } else valid = false;
      sS[rloc * TW + cn] = valid ? acc[r] : -1.0e9f;
    }
  }
  __syncthreads();

  // ---- phase 2: softmax over TW per row (16 rows x 16 threads each) ----
  const int row = tid >> 4, ti = tid & 15;
  float lmax = -3.0e38f;
  for (int c = ti; c < TW; c += 16) lmax = fmaxf(lmax, sS[row * TW + c]);
  red[tid] = lmax;
  __syncthreads();
  if (ti == 0) {
    float m = red[row * 16];
    for (int j = 1; j < 16; ++j) m = fmaxf(m, red[row * 16 + j]);
    rowmax[row] = m;
  }
  __syncthreads();
  const float rm = rowmax[row];
  float lsum = 0.f;
  for (int c = ti; c < TW; c += 16) {
    float e = __expf(sS[row * TW + c] - rm);
    sS[row * TW + c] = e;
    lsum += e;
  }
  red[tid] = lsum;
  __syncthreads();
  if (ti == 0) {
    float ssum = 0.f;
    for (int j = 0; j < 16; ++j) ssum += red[row * 16 + j];
    rowsum[row] = ssum;
  }
  __syncthreads();

  // in-place convert to f16 probabilities
  float vals[50];
#pragma unroll
  for (int i = 0; i < 50; ++i) {
    int idx = tid + i * 256;
    int r2  = idx / TW;
    vals[i] = sS[idx] / rowsum[r2];
  }
  __syncthreads();
  _Float16* sP = (_Float16*)sS;
#pragma unroll
  for (int i = 0; i < 50; ++i) sP[tid + i * 256] = (_Float16)vals[i];
  __syncthreads();

  // ---- phase 3: out[16, 64] = P[16, TW] x Vcat[TW, 64]  (waves 0..3) ----
  if (wave < 4) {
    const int dcol = h * DHD + wave * 16 + (lane & 15);
    const int mrow = lane & 15;
    vf8 acc = {0.f, 0.f, 0.f, 0.f, 0.f, 0.f, 0.f, 0.f};
    for (int kb = 0; kb < TW; kb += 32) {
      vh16 a = loadA16((const _Float16*)sP + mrow * TW + kb + koffA);
      vh16 b;
      const int kk0 = kb + koffB;
#pragma unroll
      for (int j = 0; j < 16; ++j) {
        int kk = kk0 + j;
        int tok = -1;
        if (kk < NG) tok = kk;
        else if (kk < NG + KWW) {
          int kpos = n * CHUNK + (kk - NG) - WIN;
          if (kpos >= 0 && kpos < SEQ) tok = kpos;
        }
        b[j] = (tok >= 0) ? vh[(size_t)tok * DIM + dcol] : (_Float16)0.0f;
      }
      acc = wmma32(a, b, acc);
    }
    const int rl0 = (lane >> 4) << 3;
#pragma unroll
    for (int r = 0; r < 8; ++r)
      aoh[(size_t)(qrow0 + rl0 + r) * DIM + dcol] = (_Float16)acc[r];
  }
}

// ---------------- global-token rows: full attention ----------------

__global__ __launch_bounds__(256) void ga_scores_kernel(
    const _Float16* __restrict__ qh, const _Float16* __restrict__ kh,
    float* __restrict__ sgws) {
  const int lane = threadIdx.x & 31, wave = threadIdx.x >> 5;
  const int nb = blockIdx.x;   // 256-col slab
  const int h  = blockIdx.y;
  const int koffA = (lane >> 4) << 4;
  const int koffB = (lane >> 4) << 4;
  const _Float16* aBase = qh + (size_t)(lane & 15) * DIM + h * DHD + koffA;
  const vh16 aq0 = loadA16(aBase);        // loop-invariant Q fragments
  const vh16 aq1 = loadA16(aBase + 32);
  for (int t = wave; t < 16; t += 8) {
    const int tok = nb * 256 + t * 16 + (lane & 15);
    const _Float16* bBase = kh + (size_t)tok * DIM + h * DHD + koffB;
    vf8 acc = {0.f, 0.f, 0.f, 0.f, 0.f, 0.f, 0.f, 0.f};
    acc = wmma32(aq0, load16(bBase), acc);
    acc = wmma32(aq1, load16(bBase + 32), acc);
    const int rl0 = (lane >> 4) << 3;
#pragma unroll
    for (int r = 0; r < 8; ++r)
      sgws[((size_t)h * NG + rl0 + r) * SEQ + tok] = acc[r];
  }
}

__global__ __launch_bounds__(256) void ga_softmax_kernel(
    const float* __restrict__ sgws, _Float16* __restrict__ pfh) {
  __shared__ float red[256];
  __shared__ float rowmax[16];
  __shared__ float rowsum[16];
  const int h = blockIdx.x, tid = threadIdx.x;
  const int row = tid >> 4, ti = tid & 15;
  const float* Srow = sgws + ((size_t)h * NG + row) * SEQ;
  float lmax = -3.0e38f;
  for (int c = ti; c < SEQ; c += 16) lmax = fmaxf(lmax, Srow[c]);
  red[tid] = lmax;
  __syncthreads();
  if (ti == 0) {
    float m = red[row * 16];
    for (int j = 1; j < 16; ++j) m = fmaxf(m, red[row * 16 + j]);
    rowmax[row] = m;
  }
  __syncthreads();
  const float rm = rowmax[row];
  float lsum = 0.f;
  for (int c = ti; c < SEQ; c += 16) lsum += __expf(Srow[c] - rm);
  red[tid] = lsum;
  __syncthreads();
  if (ti == 0) {
    float ssum = 0.f;
    for (int j = 0; j < 16; ++j) ssum += red[row * 16 + j];
    rowsum[row] = ssum;
  }
  __syncthreads();
  const float inv = 1.0f / rowsum[row];
  _Float16* Prow = pfh + ((size_t)h * NG + row) * SEQ;
  for (int c = ti; c < SEQ; c += 16)
    Prow[c] = (_Float16)(__expf(Srow[c] - rm) * inv);
}

__global__ __launch_bounds__(256) void ga_out_kernel(
    const _Float16* __restrict__ pfh, const _Float16* __restrict__ vh,
    _Float16* __restrict__ aoh) {
  const int lane = threadIdx.x & 31, wave = threadIdx.x >> 5;
  const int h = blockIdx.x;
  if (wave >= 4) return;
  const int dcol  = h * DHD + wave * 16 + (lane & 15);
  const int koffA = (lane >> 4) << 4;
  const int koffB = (lane >> 4) << 4;
  const _Float16* aBase = pfh + ((size_t)h * NG + (lane & 15)) * SEQ + koffA;
  vf8 acc = {0.f, 0.f, 0.f, 0.f, 0.f, 0.f, 0.f, 0.f};
  for (int kb = 0; kb < SEQ; kb += 32) {
    vh16 a = loadA16(aBase + kb);
    const int kk0 = kb + koffB;
    vh16 b;
#pragma unroll
    for (int j = 0; j < 16; ++j) b[j] = vh[(size_t)(kk0 + j) * DIM + dcol];
    acc = wmma32(a, b, acc);
  }
  const int rl0 = (lane >> 4) << 3;
#pragma unroll
  for (int r = 0; r < 8; ++r)
    aoh[(size_t)(rl0 + r) * DIM + dcol] = (_Float16)acc[r];
}

// ---------------- final mean-pool + linear ----------------

__global__ __launch_bounds__(256) void pool_fc_kernel(
    const float* __restrict__ X, const float* __restrict__ fcw,
    const float* __restrict__ fcb, float* __restrict__ out) {
  __shared__ float red[256];
  const int tid = threadIdx.x;
  float acc = 0.f;
  for (int d = tid; d < DIM; d += 256) {
    float s = 0.f;
    for (int t = 0; t < SEQ; ++t) s += X[(size_t)t * DIM + d];
    acc += (s * (1.0f / SEQ)) * fcw[d];
  }
  red[tid] = acc;
  __syncthreads();
  for (int off = 128; off > 0; off >>= 1) {
    if (tid < off) red[tid] += red[tid + off];
    __syncthreads();
  }
  if (tid == 0) out[0] = red[0] + fcb[0];
}

// ---------------- host launcher ----------------

extern "C" void kernel_launch(void* const* d_in, const int* in_sizes, int n_in,
                              void* d_out, int out_size, void* d_ws, size_t ws_size,
                              hipStream_t stream) {
  (void)in_sizes; (void)n_in; (void)out_size; (void)ws_size;
  const int*   ids    = (const int*)d_in[0];
  const float* etok   = (const float*)d_in[2];
  const float* epos   = (const float*)d_in[3];
  const float* ln_e_g = (const float*)d_in[4];
  const float* ln_e_b = (const float*)d_in[5];
  const float* wq = (const float*)d_in[6];   const float* bq = (const float*)d_in[7];
  const float* wk = (const float*)d_in[8];   const float* bk = (const float*)d_in[9];
  const float* wv = (const float*)d_in[10];  const float* bv = (const float*)d_in[11];
  const float* wo = (const float*)d_in[12];  const float* bo = (const float*)d_in[13];
  const float* ln1_g = (const float*)d_in[14]; const float* ln1_b = (const float*)d_in[15];
  const float* w1 = (const float*)d_in[16];  const float* b1 = (const float*)d_in[17];
  const float* w2 = (const float*)d_in[18];  const float* b2 = (const float*)d_in[19];
  const float* ln2_g = (const float*)d_in[20]; const float* ln2_b = (const float*)d_in[21];
  const float* fcw = (const float*)d_in[22]; const float* fcb = (const float*)d_in[23];
  float* out = (float*)d_out;

  // workspace carve
  char* p = (char*)d_ws;
  auto carve = [&](size_t bytes) {
    void* r = (void*)p;
    p += (bytes + 255) & ~(size_t)255;
    return r;
  };
  float*    X    = (float*)carve((size_t)SEQ * DIM * 4);
  float*    tmp  = (float*)carve((size_t)SEQ * DIM * 4);
  _Float16* Xh   = (_Float16*)carve((size_t)SEQ * DIM * 2);
  _Float16* qhb  = (_Float16*)carve((size_t)SEQ * DIM * 2);
  _Float16* khb  = (_Float16*)carve((size_t)SEQ * DIM * 2);
  _Float16* vhb  = (_Float16*)carve((size_t)SEQ * DIM * 2);
  _Float16* aoh  = (_Float16*)carve((size_t)SEQ * DIM * 2);
  _Float16* Hh   = (_Float16*)carve((size_t)SEQ * FF * 2);
  _Float16* wbuf = (_Float16*)carve((size_t)DIM * FF * 2);
  float*    sgws = (float*)carve((size_t)NH * NG * SEQ * 4);
  _Float16* pfh  = (_Float16*)carve((size_t)NH * NG * SEQ * 2);

  auto convT = [&](const float* W, _Float16* Wt, int K, int N) {
    int total = K * N;
    convT_kernel<<<(total + 255) / 256, 256, 0, stream>>>(W, Wt, K, N);
  };
  auto gemm = [&](const _Float16* A, const _Float16* Bt, const float* bias,
                  float* Cf, _Float16* Ch, int M, int N, int K, float alpha, int act) {
    dim3 g(N / 64, M / 128);
    wmma_gemm_kernel<<<g, 256, 0, stream>>>(A, Bt, bias, Cf, Ch, M, N, K, alpha, act);
  };

  // embeddings + LN
  embed_ln_kernel<<<SEQ, 256, 0, stream>>>(ids, etok, epos, ln_e_g, ln_e_b, X, Xh);

  for (int l = 0; l < 2; ++l) {
    const size_t wdd = (size_t)l * DIM * DIM;
    const size_t wdf = (size_t)l * DIM * FF;
    // Q, K, V projections (Q pre-scaled by 1/sqrt(DH))
    convT(wq + wdd, wbuf, DIM, DIM);
    gemm(Xh, wbuf, bq + l * DIM, nullptr, qhb, SEQ, DIM, DIM, 0.125f, 0);
    convT(wk + wdd, wbuf, DIM, DIM);
    gemm(Xh, wbuf, bk + l * DIM, nullptr, khb, SEQ, DIM, DIM, 1.0f, 0);
    convT(wv + wdd, wbuf, DIM, DIM);
    gemm(Xh, wbuf, bv + l * DIM, nullptr, vhb, SEQ, DIM, DIM, 1.0f, 0);

    // banded local + global-column attention
    attn_chunk_kernel<<<dim3(CHUNK / 16, NCH, NH), 256, 0, stream>>>(qhb, khb, vhb, aoh);
    // global-token rows (overwrite rows 0..15)
    ga_scores_kernel<<<dim3(SEQ / 256, NH), 256, 0, stream>>>(qhb, khb, sgws);
    ga_softmax_kernel<<<NH, 256, 0, stream>>>(sgws, pfh);
    ga_out_kernel<<<NH, 256, 0, stream>>>(pfh, vhb, aoh);

    // output projection + residual LN
    convT(wo + wdd, wbuf, DIM, DIM);
    gemm(aoh, wbuf, bo + l * DIM, tmp, nullptr, SEQ, DIM, DIM, 1.0f, 0);
    add_ln_kernel<<<SEQ, 256, 0, stream>>>(X, tmp, ln1_g + l * DIM, ln1_b + l * DIM, Xh);

    // FFN
    convT(w1 + wdf, wbuf, DIM, FF);
    gemm(Xh, wbuf, b1 + l * FF, nullptr, Hh, SEQ, FF, DIM, 1.0f, 1);  // GELU
    convT(w2 + wdf, wbuf, FF, DIM);
    gemm(Hh, wbuf, b2 + l * DIM, tmp, nullptr, SEQ, DIM, FF, 1.0f, 0);
    add_ln_kernel<<<SEQ, 256, 0, stream>>>(X, tmp, ln2_g + l * DIM, ln2_b + l * DIM, Xh);
  }

  pool_fc_kernel<<<1, 256, 0, stream>>>(X, fcw, fcb, out);
}
